// TGNN_48455821033656
// MI455X (gfx1250) — compile-verified, matrix operand order checked by reading, not run on hardware
//
#include <hip/hip_runtime.h>

// ---------------- Problem constants (match reference) ----------------
#define NN   100000
#define EE   300000
#define TT   3
#define BB   64
#define CIN  128
#define COUT 256

typedef __bf16 bf16_t;
typedef __attribute__((ext_vector_type(16))) __bf16 v16bf;
typedef __attribute__((ext_vector_type(8)))  float  v8f;
typedef __attribute__((ext_vector_type(2)))  int    v2i;

#if defined(__HIP_DEVICE_COMPILE__) && \
    __has_builtin(__builtin_amdgcn_global_load_async_to_lds_b64) && \
    __has_builtin(__builtin_amdgcn_s_wait_asynccnt)
#define HAS_ASYNC_LDS 1
#else
#define HAS_ASYNC_LDS 0
#endif

#define AS1 __attribute__((address_space(1)))
#define AS3 __attribute__((address_space(3)))

#if HAS_ASYNC_LDS
// Per-lane async copy of 8 bytes global -> LDS (tracked by ASYNCcnt).
__device__ __forceinline__ void async_copy8(const bf16_t* g, bf16_t* l) {
  __builtin_amdgcn_global_load_async_to_lds_b64(
      (AS1 v2i*)(AS1 void*)(void*)g,
      (AS3 v2i*)(AS3 void*)(void*)l, 0, 0);
}
#endif

__device__ __forceinline__ bf16_t f2bf(float f) {
  union { float f; unsigned u; } v; v.f = f;
  unsigned r = v.u + 0x7FFFu + ((v.u >> 16) & 1u);   // round-to-nearest-even
  unsigned short h = (unsigned short)(r >> 16);
  bf16_t o; __builtin_memcpy(&o, &h, 2); return o;
}

#define WMMA_BF16(A, Bm, C) \
  __builtin_amdgcn_wmma_f32_16x16x32_bf16(false, (A), false, (Bm), (short)0, (C), false, false)

// ---------------- Utility kernels ----------------
__global__ void zero_k(float* __restrict__ p, long n) {
  long i = (long)blockIdx.x * blockDim.x + threadIdx.x;
  long s = (long)gridDim.x * blockDim.x;
  for (; i < n; i += s) p[i] = 0.0f;
}

__global__ void f2bf_arr(bf16_t* __restrict__ dst, const float* __restrict__ src, long n) {
  long i = (long)blockIdx.x * blockDim.x + threadIdx.x;
  long s = (long)gridDim.x * blockDim.x;
  for (; i < n; i += s) dst[i] = f2bf(src[i]);
}

// ---------------- Linear: Y = A @ W^T + b  (A bf16 [N,K], W bf16 [256,K]) ----
// 128 threads = 4 waves; block covers 16 rows x 64 cols; wave -> 16x16 tile.
template<int K>
__global__ __launch_bounds__(128) void lin_gemm(const bf16_t* __restrict__ A,
                                                const bf16_t* __restrict__ Wb,
                                                const float*  __restrict__ bias,
                                                float*  __restrict__ Yf,
                                                bf16_t* __restrict__ Yb) {
  __shared__ __attribute__((aligned(32))) bf16_t lA[2][16 * 32];
  const int tid = threadIdx.x, lane = tid & 31, wave = tid >> 5;
  const int rowTile = blockIdx.x >> 2, colTile = blockIdx.x & 3;
  const int row0 = rowTile * 16;
  const int col0 = colTile * 64 + wave * 16;
  const int m = lane & 15, kh = lane >> 4;
  const int r = tid >> 3, kk = (tid & 7) * 4;            // staging: 8B per thread
  const bf16_t* gA = A + (size_t)(row0 + r) * K + kk;
  v8f acc = {};
#if HAS_ASYNC_LDS
  // Double-buffered async global->LDS pipeline.
  async_copy8(gA, &lA[0][r * 32 + kk]);
  for (int kc = 0; kc < K; kc += 32) {
    const int cur = (kc >> 5) & 1;
    if (kc + 32 < K) {
      async_copy8(gA + kc + 32, &lA[1 - cur][r * 32 + kk]);
      __builtin_amdgcn_s_wait_asynccnt(1);               // current tile landed
    } else {
      __builtin_amdgcn_s_wait_asynccnt(0);
    }
    __syncthreads();
    v16bf a = *(const v16bf*)&lA[cur][m * 32 + kh * 16];
    v16bf b = *(const v16bf*)(Wb + (size_t)(col0 + m) * K + kc + kh * 16);
    acc = WMMA_BF16(a, b, acc);
    __syncthreads();                                     // reads done before rewrite
  }
#else
  for (int kc = 0; kc < K; kc += 32) {
    __syncthreads();
    ((uint2*)lA[0])[tid] = *(const uint2*)(gA + kc);
    if (kc + 32 < K) __builtin_prefetch(gA + kc + 32, 0, 1);
    __syncthreads();
    v16bf a = *(const v16bf*)&lA[0][m * 32 + kh * 16];
    v16bf b = *(const v16bf*)(Wb + (size_t)(col0 + m) * K + kc + kh * 16);
    acc = WMMA_BF16(a, b, acc);
  }
#endif
  const int col = col0 + m;
  const float bv = bias[col];
#pragma unroll
  for (int i = 0; i < 8; ++i) {
    const int row = row0 + i + kh * 8;            // C layout: lanes 16-31 -> M+8
    const size_t idx = (size_t)row * COUT + col;
    const float y = acc[i] + bv;
    Yf[idx] = y;
    Yb[idx] = f2bf(y);
  }
}

// ---------------- Fused GRU cell (gi/gh GEMMs + gates + where) ---------------
__global__ __launch_bounds__(128) void gru_kernel(const bf16_t* __restrict__ Xb,
                                                  const bf16_t* __restrict__ Hb,
                                                  const bf16_t* __restrict__ Wih,
                                                  const bf16_t* __restrict__ Whh,
                                                  const float*  __restrict__ bih,
                                                  const float*  __restrict__ bhh,
                                                  float*  __restrict__ Xf,
                                                  const float*  __restrict__ Hf,
                                                  bf16_t* __restrict__ XbOut) {
  __shared__ __attribute__((aligned(32))) bf16_t lx[2][16 * 32];
  __shared__ __attribute__((aligned(32))) bf16_t lh[2][16 * 32];
  const int tid = threadIdx.x, lane = tid & 31, wave = tid >> 5;
  const int rowTile = blockIdx.x >> 2, colTile = blockIdx.x & 3;
  const int row0 = rowTile * 16;
  const int col0 = colTile * 64 + wave * 16;
  const int m = lane & 15, kh = lane >> 4;
  const int K = COUT;
  const int r0 = tid >> 3, kk = (tid & 7) * 4;           // staging: 8B per thread
  const bf16_t* gx = Xb + (size_t)(row0 + r0) * K + kk;
  const bf16_t* gh = Hb + (size_t)(row0 + r0) * K + kk;
  v8f ar = {}, az = {}, ai = {}, ah = {};
#if HAS_ASYNC_LDS
  async_copy8(gx, &lx[0][r0 * 32 + kk]);
  async_copy8(gh, &lh[0][r0 * 32 + kk]);
#endif
  for (int kc = 0; kc < K; kc += 32) {
#if HAS_ASYNC_LDS
    const int cur = (kc >> 5) & 1;
    if (kc + 32 < K) {
      async_copy8(gx + kc + 32, &lx[1 - cur][r0 * 32 + kk]);
      async_copy8(gh + kc + 32, &lh[1 - cur][r0 * 32 + kk]);
      __builtin_amdgcn_s_wait_asynccnt(2);               // current tile landed
    } else {
      __builtin_amdgcn_s_wait_asynccnt(0);
    }
    __syncthreads();
    v16bf ax  = *(const v16bf*)&lx[cur][m * 32 + kh * 16];
    v16bf axh = *(const v16bf*)&lh[cur][m * 32 + kh * 16];
#else
    __syncthreads();
    ((uint2*)lx[0])[tid] = *(const uint2*)(gx + kc);
    ((uint2*)lh[0])[tid] = *(const uint2*)(gh + kc);
    if (kc + 32 < K) {
      __builtin_prefetch(gx + kc + 32, 0, 1);
      __builtin_prefetch(gh + kc + 32, 0, 1);
    }
    __syncthreads();
    v16bf ax  = *(const v16bf*)&lx[0][m * 32 + kh * 16];
    v16bf axh = *(const v16bf*)&lh[0][m * 32 + kh * 16];
#endif
    const size_t wro = (size_t)(col0 + m) * K + kc + kh * 16;  // r-gate row
    v16bf b;
    b = *(const v16bf*)(Wih + wro);                    ar = WMMA_BF16(ax,  b, ar);
    b = *(const v16bf*)(Whh + wro);                    ar = WMMA_BF16(axh, b, ar);
    b = *(const v16bf*)(Wih + wro + (size_t)256 * K);  az = WMMA_BF16(ax,  b, az);
    b = *(const v16bf*)(Whh + wro + (size_t)256 * K);  az = WMMA_BF16(axh, b, az);
    b = *(const v16bf*)(Wih + wro + (size_t)512 * K);  ai = WMMA_BF16(ax,  b, ai);
    b = *(const v16bf*)(Whh + wro + (size_t)512 * K);  ah = WMMA_BF16(axh, b, ah);
#if HAS_ASYNC_LDS
    __syncthreads();                                   // reads done before rewrite
#endif
  }
  const int col = col0 + m;
  const float br  = bih[col] + bhh[col];
  const float bz  = bih[256 + col] + bhh[256 + col];
  const float bni = bih[512 + col];
  const float bnh = bhh[512 + col];
#pragma unroll
  for (int i = 0; i < 8; ++i) {
    const int row = row0 + i + kh * 8;
    const size_t idx = (size_t)row * COUT + col;
    const float r = 1.0f / (1.0f + __expf(-(ar[i] + br)));
    const float z = 1.0f / (1.0f + __expf(-(az[i] + bz)));
    const float n = tanhf(ai[i] + bni + r * (ah[i] + bnh));
    const float hv = Hf[idx];
    const float xv = Xf[idx];
    const float hhat = (1.0f - z) * n + z * hv;
    const float o = (hv == 0.0f) ? xv : hhat;     // update(): passthrough where h==0
    Xf[idx] = o;
    XbOut[idx] = f2bf(o);
  }
}

// ---------------- Edge scatter-sum (one block per edge) ----------------
__global__ __launch_bounds__(256) void edge_scatter(const int* __restrict__ src,
                                                    const int* __restrict__ dst,
                                                    const float* __restrict__ Xf,
                                                    float* __restrict__ Hsum,
                                                    float* __restrict__ Cnt) {
  const int e = blockIdx.x, f = threadIdx.x;
  const int s = src[e], d = dst[e];
  atomicAdd(&Hsum[(size_t)d * COUT + f], Xf[(size_t)s * COUT + f]);
  if (f == 0) atomicAdd(&Cnt[d], 1.0f);
}

__global__ __launch_bounds__(256) void finalize_h(float* __restrict__ H,
                                                  const float* __restrict__ Cnt,
                                                  bf16_t* __restrict__ Hb) {
  const int nidx = blockIdx.x, f = threadIdx.x;
  float c = Cnt[nidx]; c = c < 1.0f ? 1.0f : c;
  const size_t idx = (size_t)nidx * COUT + f;
  const float h = H[idx] / c;
  H[idx] = h;
  Hb[idx] = f2bf(h);
}

__global__ void relu_k(float* __restrict__ Xf, bf16_t* __restrict__ Xb, long n) {
  long i = (long)blockIdx.x * blockDim.x + threadIdx.x;
  long s = (long)gridDim.x * blockDim.x;
  for (; i < n; i += s) {
    float v = Xf[i]; v = v > 0.0f ? v : 0.0f;
    Xf[i] = v; Xb[i] = f2bf(v);
  }
}

// ---------------- Graph mean pool ----------------
__global__ __launch_bounds__(256) void pool_scatter(const float* __restrict__ Xf,
                                                    const int* __restrict__ offs,
                                                    float* __restrict__ Out,
                                                    float* __restrict__ Gc) {
  const int nidx = blockIdx.x, f = threadIdx.x;
  const int g = offs[nidx];
  atomicAdd(&Out[(size_t)g * COUT + f], Xf[(size_t)nidx * COUT + f]);
  if (f == 0) atomicAdd(&Gc[g], 1.0f);
}

__global__ void pool_div(float* __restrict__ Out, const float* __restrict__ Gc) {
  const int i = blockIdx.x * blockDim.x + threadIdx.x;
  if (i < BB * COUT) {
    float c = Gc[i >> 8]; c = c < 1.0f ? 1.0f : c;
    Out[i] /= c;
  }
}

// ---------------- Host orchestration ----------------
extern "C" void kernel_launch(void* const* d_in, const int* in_sizes, int n_in,
                              void* d_out, int out_size, void* d_ws, size_t ws_size,
                              hipStream_t stream) {
  const float* x      = (const float*)d_in[0];
  const int*   offset = (const int*)  d_in[1];
  const int*   edge   = (const int*)  d_in[2];   // [T,2,E]
  const float* W1     = (const float*)d_in[3];
  const float* b1     = (const float*)d_in[4];
  const float* W2     = (const float*)d_in[5];
  const float* b2     = (const float*)d_in[6];
  const float* Wih    = (const float*)d_in[7];
  const float* Whh    = (const float*)d_in[8];
  const float* bih    = (const float*)d_in[9];
  const float* bhh    = (const float*)d_in[10];
  float* out = (float*)d_out;

  char* wsp = (char*)d_ws;
  auto alloc = [&](size_t bytes) -> char* {
    char* p = wsp; wsp += (bytes + 255) & ~(size_t)255; return p;
  };
  bf16_t* w1b  = (bf16_t*)alloc((size_t)COUT * CIN * 2);
  bf16_t* w2b  = (bf16_t*)alloc((size_t)COUT * COUT * 2);
  bf16_t* wihb = (bf16_t*)alloc((size_t)3 * COUT * COUT * 2);
  bf16_t* whhb = (bf16_t*)alloc((size_t)3 * COUT * COUT * 2);
  bf16_t* xinb = (bf16_t*)alloc((size_t)NN * CIN * 2);
  float*  xf   = (float*) alloc((size_t)NN * COUT * 4);
  bf16_t* xb0  = (bf16_t*)alloc((size_t)NN * COUT * 2);
  bf16_t* xb1  = (bf16_t*)alloc((size_t)NN * COUT * 2);
  float*  hf   = (float*) alloc((size_t)NN * COUT * 4);   // sum, then mean in place
  bf16_t* hb   = (bf16_t*)alloc((size_t)NN * COUT * 2);
  float*  cnt  = (float*) alloc((size_t)NN * 4);
  float*  gcnt = (float*) alloc((size_t)BB * 4);
  bf16_t* xb[2] = {xb0, xb1};

  const int GEMM_GRID = (NN / 16) * 4;   // 6250 row tiles x 4 col tiles

  // Weight / input conversion to bf16 (stays L2-resident)
  f2bf_arr<<<256, 256, 0, stream>>>(w1b,  W1,  (long)COUT * CIN);
  f2bf_arr<<<256, 256, 0, stream>>>(w2b,  W2,  (long)COUT * COUT);
  f2bf_arr<<<512, 256, 0, stream>>>(wihb, Wih, (long)3 * COUT * COUT);
  f2bf_arr<<<512, 256, 0, stream>>>(whhb, Whh, (long)3 * COUT * COUT);
  f2bf_arr<<<2048, 256, 0, stream>>>(xinb, x,  (long)NN * CIN);

  int cur = 0;
  for (int layer = 0; layer < 2; ++layer) {
    if (layer == 0) {
      lin_gemm<CIN><<<GEMM_GRID, 128, 0, stream>>>(xinb, w1b, b1, xf, xb[0]);
      cur = 0;
    } else {
      lin_gemm<COUT><<<GEMM_GRID, 128, 0, stream>>>(xb[cur], w2b, b2, xf, xb[1 - cur]);
      cur ^= 1;
    }
    for (int t = 0; t < TT; ++t) {
      zero_k<<<2048, 256, 0, stream>>>(hf, (long)NN * COUT);
      zero_k<<<256, 256, 0, stream>>>(cnt, (long)NN);
      const int* src = edge + (size_t)t * 2 * EE;
      const int* dst = src + EE;
      edge_scatter<<<EE, 256, 0, stream>>>(src, dst, xf, hf, cnt);
      finalize_h<<<NN, 256, 0, stream>>>(hf, cnt, hb);
      gru_kernel<<<GEMM_GRID, 128, 0, stream>>>(xb[cur], hb, wihb, whhb,
                                                bih, bhh, xf, hf, xb[1 - cur]);
      cur ^= 1;
    }
    relu_k<<<2048, 256, 0, stream>>>(xf, xb[cur], (long)NN * COUT);
  }

  // Global mean pool over graph ids
  zero_k<<<32, 256, 0, stream>>>(out, (long)BB * COUT);
  zero_k<<<1, 64, 0, stream>>>(gcnt, (long)BB);
  pool_scatter<<<NN, 256, 0, stream>>>(xf, offset, out, gcnt);
  pool_div<<<64, 256, 0, stream>>>(out, gcnt);
}
